// VGG_SNN_MC_84550726189408
// MI455X (gfx1250) — matrix operand order repaired and checked by reading, not compiled, not run on hardware
//
#include <hip/hip_runtime.h>

typedef __attribute__((ext_vector_type(16))) _Float16 v16h;
typedef __attribute__((ext_vector_type(8)))  _Float16 h8;
typedef __attribute__((ext_vector_type(8)))  float    v8f;
typedef __attribute__((ext_vector_type(4)))  float    f4;

#define SNN_B      4
#define SNN_T      3
#define SNN_LABELS 10

// ---- async global->LDS copy (gfx1250 GLOBAL_LOAD_ASYNC_TO_LDS_B128) --------
#if defined(__has_builtin)
#if __has_builtin(__builtin_amdgcn_global_load_async_to_lds_b128)
#define HAVE_ASYNC_LDS 1
#endif
#endif

#if defined(HAVE_ASYNC_LDS)
typedef __attribute__((__vector_size__(4 * sizeof(int)))) int v4i_ld;
typedef __attribute__((address_space(1))) v4i_ld* g_v4i_p;
typedef __attribute__((address_space(3))) v4i_ld* l_v4i_p;
__device__ __forceinline__ void async_copy_b128(void* lds, const void* g) {
  __builtin_amdgcn_global_load_async_to_lds_b128(
      (g_v4i_p)(g), (l_v4i_p)(lds), 0, 0);
}
__device__ __forceinline__ void async_wait0() {
#if __has_builtin(__builtin_amdgcn_s_wait_asynccnt)
  __builtin_amdgcn_s_wait_asynccnt(0);
#else
  asm volatile("s_wait_asynccnt 0x0" ::: "memory");
#endif
}
#endif

// ---------------------------------------------------------------------------
// LIF multi-compartment spike (K=2), division-free:
// compartment i fires iff 2^i*thr < m <= 2^{i+1}*thr  (thr > 0)
// ---------------------------------------------------------------------------
__device__ __forceinline__ float spike_val(float m, float thr, float t2, float t4) {
  float val = 0.0f;
  val = (m > thr && m <= t2) ? 1.0f : val;
  val = (m > t2  && m <= t4) ? 2.0f : val;
  return val;
}

// 8-element fused LIF epilogue: mem/act at `base`, contiguous channels.
__device__ __forceinline__ void lif_store8(const v8f& acc, float* __restrict__ mem,
                                           _Float16* __restrict__ act, size_t base,
                                           float thr, float leak) {
  const float t2 = 2.0f * thr, t4 = 4.0f * thr;
  f4 m0 = *(const f4*)(mem + base);
  f4 m1 = *(const f4*)(mem + base + 4);
  h8 vout;
  f4 mo0, mo1;
#pragma unroll
  for (int i = 0; i < 4; ++i) {
    float ma = leak * m0[i] + acc[i];
    float va = spike_val(ma, thr, t2, t4);
    mo0[i]  = ma - thr * va;
    vout[i] = (_Float16)va;
    float mb = leak * m1[i] + acc[4 + i];
    float vb = spike_val(mb, thr, t2, t4);
    mo1[i]      = mb - thr * vb;
    vout[4 + i] = (_Float16)vb;
  }
  *(f4*)(mem + base)     = mo0;
  *(f4*)(mem + base + 4) = mo1;
  *(h8*)(act + base)     = vout;
}

// ---------------------------------------------------------------------------
// Utility / prepack kernels (launch-invariant; rerun every call)
// ---------------------------------------------------------------------------
__global__ void zero_f32(float* __restrict__ p, int n) {
  int i = blockIdx.x * blockDim.x + threadIdx.x;
  if (i < n) p[i] = 0.0f;
}

__global__ void cvt_f32_f16(const float* __restrict__ src,
                            _Float16* __restrict__ dst, int n) {
  int i = blockIdx.x * blockDim.x + threadIdx.x;
  if (i < n) dst[i] = (_Float16)src[i];
}

// conv weights fp32 [Cout][Cin][3][3] -> f16 [Cout][9][Cin]  (k = r*Cin + ci)
__global__ void pack_conv_w(const float* __restrict__ src,
                            _Float16* __restrict__ dst, int Cout, int Cin) {
  int i = blockIdx.x * blockDim.x + threadIdx.x;
  int total = Cout * Cin * 9;
  if (i >= total) return;
  int ci = i % Cin;
  int t  = i / Cin;
  int r  = t % 9;
  int co = t / 9;
  dst[i] = (_Float16)src[((size_t)co * Cin + ci) * 9 + r];
}

// conv0 weights fp32 [64][3][3][3] -> f16 [64][32], k = r*3+ci, zero padded
__global__ void pack_conv0_w(const float* __restrict__ src,
                             _Float16* __restrict__ dst) {
  int i = blockIdx.x * blockDim.x + threadIdx.x;
  if (i >= 64 * 32) return;
  int k  = i & 31;
  int co = i >> 5;
  _Float16 v = (_Float16)0.0f;
  if (k < 27) {
    int rr = k / 3, ci = k - rr * 3;
    v = (_Float16)src[((size_t)co * 3 + ci) * 9 + rr];
  }
  dst[i] = v;
}

// fc0 fp32 [4096][2048] (f = c*4 + h*2 + w) -> f16 with k' = (h*2+w)*512 + c,
// so B columns are contiguous slices of the channel-last conv12 buffer.
__global__ void pack_fc0_w(const float* __restrict__ src,
                           _Float16* __restrict__ dst) {
  int i = blockIdx.x * blockDim.x + threadIdx.x;
  if (i >= 4096 * 2048) return;
  int kp = i & 2047;
  int o  = i >> 11;
  int c  = kp & 511;
  int p  = kp >> 9;
  dst[i] = (_Float16)src[(size_t)o * 2048 + (c * 4 + p)];
}

// x [B][3][32*32] fp32 -> channel-last f16 [B][32][32][3]
__global__ void input_to_cl_f16(const float* __restrict__ x,
                                _Float16* __restrict__ dst, int Bn) {
  int i = blockIdx.x * blockDim.x + threadIdx.x;
  int total = Bn * 1024 * 3;
  if (i >= total) return;
  int c = i % 3;
  int p = (i / 3) % 1024;
  int b = i / (3 * 1024);
  dst[i] = (_Float16)x[((size_t)b * 3 + c) * 1024 + p];
}

// 2x2 avg pool, channel-last [B][S][S][C] -> [B][S/2][S/2][C]
__global__ void avgpool2_cl_f16(const _Float16* __restrict__ in,
                                _Float16* __restrict__ out,
                                int C, int Bn, int S) {
  int i = blockIdx.x * blockDim.x + threadIdx.x;
  int So = S >> 1;
  int total = Bn * So * So * C;
  if (i >= total) return;
  int c   = i % C;
  int pix = i / C;
  int w   = pix % So;
  pix /= So;
  int h = pix % So;
  int b = pix / So;
  const _Float16* p00 = in + ((size_t)(b * S + 2 * h) * S + 2 * w) * C + c;
  float s = (float)p00[0] + (float)p00[C] +
            (float)p00[(size_t)S * C] + (float)p00[(size_t)S * C + C];
  out[i] = (_Float16)(0.25f * s);
}

// ---------------------------------------------------------------------------
// conv0: Cin=3, K padded to 32, one wmma per tile, branchless clamped gather.
// ---------------------------------------------------------------------------
__global__ __launch_bounds__(128)
void conv0_lif_wmma(const _Float16* __restrict__ Wf,   // [64][32]
                    const _Float16* __restrict__ X,    // [B][32][32][3]
                    _Float16* __restrict__ actOut,     // [B][32][32][64]
                    float* __restrict__ mem,
                    const float* __restrict__ thrArr,
                    const float* __restrict__ leakArr, int Bn) {
  const int lane  = threadIdx.x & 31;
  const int wid   = threadIdx.x >> 5;
  const int lhalf = lane >> 4;
  const int l16   = lane & 15;
  const int Cout  = 64, S = 32, HW = 1024;
  const int mtile = blockIdx.y * 4 + wid;

  const int n = blockIdx.x * 16 + l16;
  const int b = n / HW;
  int rrem    = n - b * HW;
  const int h = rrem / S;
  const int w = rrem - h * S;

  const _Float16* Arow = Wf + (size_t)(mtile * 16 + l16) * 32;
  const h8 alo = *(const h8*)(Arow + lhalf * 16);
  const h8 ahi = *(const h8*)(Arow + lhalf * 16 + 8);
  v16h a = __builtin_shufflevector(alo, ahi, 0, 1, 2, 3, 4, 5, 6, 7,
                                   8, 9, 10, 11, 12, 13, 14, 15);
  v16h bf;
#pragma unroll
  for (int j = 0; j < 16; ++j) {
    int k  = lhalf * 16 + j;
    int rr = k / 3, ci = k - rr * 3;
    int kh = rr / 3, kw = rr - kh * 3;
    int hh = h + kh - 1, ww = w + kw - 1;
    bool ok = (k < 27) & (hh >= 0) & (hh < S) & (ww >= 0) & (ww < S);
    int hc = hh < 0 ? 0 : (hh > 31 ? 31 : hh);
    int wc = ww < 0 ? 0 : (ww > 31 ? 31 : ww);
    _Float16 v = X[((size_t)(b * S + hc) * S + wc) * 3 + ci];
    bf[j] = ok ? v : (_Float16)0.0f;
  }
  v8f acc = {};
  acc = __builtin_amdgcn_wmma_f32_16x16x32_f16(false, a, false, bf,
                                               (short)0, acc, false, false);

  size_t base = (size_t)n * Cout + mtile * 16 + lhalf * 8;
  lif_store8(acc, mem, actOut, base, thrArr[0], leakArr[0]);
}

// ---------------------------------------------------------------------------
// Generic 3x3 SAME conv + fused LIF, channel-last, CIN % 32 == 0.
// Block = 4 waves covering 4 M tiles x ONE shared 32-column N slab:
//   - B pixels staged once per (r) in LDS (async global->LDS when available,
//     else load+ds_store), halo columns sourced from a zeroed global pixel;
//   - each wave holds TWO N-tile accumulators sharing one A fragment.
// ---------------------------------------------------------------------------
template <int CIN>
__global__ __launch_bounds__(128)
void conv_lif_wmma(const _Float16* __restrict__ Wf,   // [Cout][9][CIN]
                   const _Float16* __restrict__ actIn,
                   const _Float16* __restrict__ zeroPix, // >= CIN zero halfs
                   _Float16* __restrict__ actOut,
                   float* __restrict__ mem,
                   const float* __restrict__ thrArr,
                   const float* __restrict__ leakArr,
                   int layer, int Cout, int S, int Bn) {
  __shared__ _Float16 Btile[32 * CIN];

  const int tid   = threadIdx.x;
  const int lane  = tid & 31;
  const int wid   = tid >> 5;
  const int lhalf = (lane >> 4) & 1;
  const int l16   = lane & 15;
  const int HW    = S * S;
  const int NP    = Bn * HW;
  const int mtile = blockIdx.y * 4 + wid;
  const int n0    = blockIdx.x * 32;

  // fill assignment: 4 threads per LDS column
  const int fcol = tid >> 2;
  const int fsub = tid & 3;
  int nf = n0 + fcol;
  const bool fcol_v = nf < NP;
  if (!fcol_v) nf = NP - 1;
  const int fb = nf / HW;
  int frr      = nf - fb * HW;
  const int fh = frr / S;
  const int fw = frr - fh * S;

  // output columns for this lane (tile0 always fully valid; tile1 uniform flag)
  const int  na  = n0 + l16;
  const int  nb  = n0 + 16 + l16;
  const bool t1v = (n0 + 16) < NP;

  const _Float16* Arow = Wf + (size_t)(mtile * 16 + l16) * 9 * CIN;

  v8f acc0 = {}, acc1 = {};
#pragma unroll
  for (int r = 0; r < 9; ++r) {
    const int kh = r / 3, kw = r - kh * 3;
    // ---- cooperative B-slab fill (global -> LDS), halo -> zeroPix ----
    {
      int hh = fh + kh - 1, ww = fw + kw - 1;
      bool v = fcol_v & (hh >= 0) & (hh < S) & (ww >= 0) & (ww < S);
      const _Float16* pix =
          v ? (actIn + ((size_t)(fb * S + hh) * S + ww) * CIN) : zeroPix;
#if defined(HAVE_ASYNC_LDS)
#pragma unroll
      for (int c0 = fsub * 16; c0 < CIN; c0 += 64) {
        async_copy_b128(&Btile[fcol * CIN + c0],     pix + c0);
        async_copy_b128(&Btile[fcol * CIN + c0 + 8], pix + c0 + 8);
      }
      async_wait0();
#else
#pragma unroll
      for (int c0 = fsub * 16; c0 < CIN; c0 += 64) {
        h8 d0 = *(const h8*)(pix + c0);
        h8 d1 = *(const h8*)(pix + c0 + 8);
        *(h8*)(&Btile[fcol * CIN + c0])     = d0;
        *(h8*)(&Btile[fcol * CIN + c0 + 8]) = d1;
      }
#endif
    }
    __syncthreads();
    // ---- 2 N-tiles per wave, A fragment reused across both wmmas ----
    const _Float16* wr = Arow + r * CIN;
#pragma unroll 4
    for (int c0 = 0; c0 < CIN; c0 += 32) {
      const h8 alo = *(const h8*)(wr + c0 + lhalf * 16);
      const h8 ahi = *(const h8*)(wr + c0 + lhalf * 16 + 8);
      v16h a = __builtin_shufflevector(alo, ahi, 0, 1, 2, 3, 4, 5, 6, 7,
                                       8, 9, 10, 11, 12, 13, 14, 15);
      const _Float16* bp0 = &Btile[(size_t)l16 * CIN + c0 + lhalf * 16];
      const _Float16* bp1 = &Btile[(size_t)(16 + l16) * CIN + c0 + lhalf * 16];
      v16h b0 = __builtin_shufflevector(*(const h8*)bp0, *(const h8*)(bp0 + 8),
                                        0, 1, 2, 3, 4, 5, 6, 7,
                                        8, 9, 10, 11, 12, 13, 14, 15);
      v16h b1 = __builtin_shufflevector(*(const h8*)bp1, *(const h8*)(bp1 + 8),
                                        0, 1, 2, 3, 4, 5, 6, 7,
                                        8, 9, 10, 11, 12, 13, 14, 15);
      acc0 = __builtin_amdgcn_wmma_f32_16x16x32_f16(false, a, false, b0,
                                                    (short)0, acc0, false, false);
      acc1 = __builtin_amdgcn_wmma_f32_16x16x32_f16(false, a, false, b1,
                                                    (short)0, acc1, false, false);
    }
    __syncthreads();
  }

  const float thr  = thrArr[layer];
  const float leak = leakArr[layer];
  const int   cbase = mtile * 16 + lhalf * 8;
  lif_store8(acc0, mem, actOut, (size_t)na * Cout + cbase, thr, leak);
  if (t1v)
    lif_store8(acc1, mem, actOut, (size_t)nb * Cout + cbase, thr, leak);
}

// ---------------------------------------------------------------------------
// FC layer + fused LIF (fc0/fc1): prepacked f16 weights [OutF][InF];
// B column b = InF contiguous halfs. Outputs batch-major [b][OutF].
// ---------------------------------------------------------------------------
__global__ __launch_bounds__(128)
void fc_lif_wmma(const _Float16* __restrict__ Wp,
                 const _Float16* __restrict__ actIn,
                 _Float16* __restrict__ actOut,
                 float* __restrict__ mem,
                 const float* __restrict__ thrArr,
                 const float* __restrict__ leakArr,
                 int layer, int InF, int OutF, int Bn) {
  const int lane  = threadIdx.x & 31;
  const int wid   = threadIdx.x >> 5;
  const int lhalf = lane >> 4;
  const int l16   = lane & 15;
  const int mtile = blockIdx.y * 4 + wid;

  const int n  = l16;
  const int bb = n < Bn ? n : 0;
  const _Float16* Arow = Wp + (size_t)(mtile * 16 + l16) * InF;
  const _Float16* Bcol = actIn + (size_t)bb * InF;

  v8f acc = {};
  for (int k0 = 0; k0 < InF; k0 += 32) {
    if (k0 + 1024 < InF) __builtin_prefetch(Arow + k0 + 1024, 0, 1);
    const h8 alo = *(const h8*)(Arow + k0 + lhalf * 16);
    const h8 ahi = *(const h8*)(Arow + k0 + lhalf * 16 + 8);
    const h8 b0  = *(const h8*)(Bcol + k0 + lhalf * 16);
    const h8 b1  = *(const h8*)(Bcol + k0 + lhalf * 16 + 8);
    v16h a  = __builtin_shufflevector(alo, ahi, 0, 1, 2, 3, 4, 5, 6, 7,
                                      8, 9, 10, 11, 12, 13, 14, 15);
    v16h bf = __builtin_shufflevector(b0, b1, 0, 1, 2, 3, 4, 5, 6, 7,
                                      8, 9, 10, 11, 12, 13, 14, 15);
    acc = __builtin_amdgcn_wmma_f32_16x16x32_f16(false, a, false, bf,
                                                 (short)0, acc, false, false);
  }

  if (n < Bn) {
    size_t base = (size_t)n * OutF + mtile * 16 + lhalf * 8;
    lif_store8(acc, mem, actOut, base, thrArr[layer], leakArr[layer]);
  }
}

// ---------------------------------------------------------------------------
// fc2 classifier (prepacked f16 [10][4096] padded rows clamped):
// logits[b][l] += act @ fc2^T; accumulates across the 3 timesteps.
// ---------------------------------------------------------------------------
__global__ __launch_bounds__(32)
void fc2_logits_wmma(const _Float16* __restrict__ Wp,
                     const _Float16* __restrict__ actIn,
                     float* __restrict__ logits,
                     int InF, int Labels, int Bn) {
  const int lane  = threadIdx.x & 31;
  const int lhalf = lane >> 4;
  const int l16   = lane & 15;
  const int n  = l16;
  const int bb = n < Bn ? n : 0;
  const int rc = l16 < Labels ? l16 : 0;
  const _Float16* Arow = Wp + (size_t)rc * InF;
  const _Float16* Bcol = actIn + (size_t)bb * InF;

  v8f acc = {};
  for (int k0 = 0; k0 < InF; k0 += 32) {
    const h8 alo = *(const h8*)(Arow + k0 + lhalf * 16);
    const h8 ahi = *(const h8*)(Arow + k0 + lhalf * 16 + 8);
    const h8 b0  = *(const h8*)(Bcol + k0 + lhalf * 16);
    const h8 b1  = *(const h8*)(Bcol + k0 + lhalf * 16 + 8);
    v16h a  = __builtin_shufflevector(alo, ahi, 0, 1, 2, 3, 4, 5, 6, 7,
                                      8, 9, 10, 11, 12, 13, 14, 15);
    v16h bf = __builtin_shufflevector(b0, b1, 0, 1, 2, 3, 4, 5, 6, 7,
                                      8, 9, 10, 11, 12, 13, 14, 15);
    acc = __builtin_amdgcn_wmma_f32_16x16x32_f16(false, a, false, bf,
                                                 (short)0, acc, false, false);
  }
#pragma unroll
  for (int i = 0; i < 8; ++i) {
    int M = i + lhalf * 8;
    if (M < Labels && n < Bn) logits[(size_t)n * Labels + M] += acc[i];
  }
}

// ---------------------------------------------------------------------------
// Host orchestration (graph-capture safe: launches only, deterministic).
// ---------------------------------------------------------------------------
extern "C" void kernel_launch(void* const* d_in, const int* in_sizes, int n_in,
                              void* d_out, int out_size, void* d_ws, size_t ws_size,
                              hipStream_t stream) {
  (void)in_sizes; (void)n_in; (void)ws_size;

  const float* x = (const float*)d_in[0];
  const float* wconv[13];
  for (int i = 0; i < 13; ++i) wconv[i] = (const float*)d_in[1 + i];
  const float* fc0w = (const float*)d_in[14];
  const float* fc1w = (const float*)d_in[15];
  const float* fc2w = (const float*)d_in[16];
  const float* thr  = (const float*)d_in[17];
  const float* leak = (const float*)d_in[18];
  float*       out  = (float*)d_out;

  const int Bn = SNN_B;
  static const int cfg[13] = {64, 64, 128, 128, 256, 256, 256,
                              512, 512, 512, 512, 512, 512};
  int convS[13];
  {
    int s = 32;
    for (int i = 0; i < 13; ++i) {
      convS[i] = s;
      if (i == 1 || i == 3 || i == 6 || i == 9) s >>= 1;
    }
  }

  // ---- carve workspace (256B-aligned) ----
  char*  ws  = (char*)d_ws;
  size_t off = 0;
  auto carve = [&](size_t bytes) -> char* {
    char* p = ws + off;
    off += (bytes + 255) & ~(size_t)255;
    return p;
  };
  _Float16* zeroPix = (_Float16*)carve(512 * sizeof(_Float16));
  _Float16* Xf16 = (_Float16*)carve((size_t)Bn * 1024 * 3 * sizeof(_Float16));
  _Float16* actA = (_Float16*)carve((size_t)Bn * 1024 * 64 * sizeof(_Float16));
  _Float16* actB = (_Float16*)carve((size_t)Bn * 1024 * 64 * sizeof(_Float16));
  _Float16* fcA  = (_Float16*)carve((size_t)Bn * 4096 * sizeof(_Float16));
  _Float16* fcB  = (_Float16*)carve((size_t)Bn * 4096 * sizeof(_Float16));

  size_t memElems[15];
  size_t memTotal = 0;
  for (int i = 0; i < 13; ++i) {
    memElems[i] = (size_t)cfg[i] * Bn * convS[i] * convS[i];
    memTotal += memElems[i];
  }
  memElems[13] = (size_t)4096 * Bn;
  memElems[14] = (size_t)4096 * Bn;
  memTotal += memElems[13] + memElems[14];
  float* memBase = (float*)carve(memTotal * sizeof(float));
  float* memPtr[15];
  {
    size_t o = 0;
    for (int i = 0; i < 15; ++i) { memPtr[i] = memBase + o; o += memElems[i]; }
  }

  // prepacked f16 weights
  _Float16* wpack[13];
  wpack[0] = (_Float16*)carve((size_t)64 * 32 * sizeof(_Float16));
  for (int i = 1; i < 13; ++i)
    wpack[i] = (_Float16*)carve((size_t)cfg[i] * 9 * cfg[i - 1] * sizeof(_Float16));
  _Float16* fc0p = (_Float16*)carve((size_t)4096 * 2048 * sizeof(_Float16));
  _Float16* fc1p = (_Float16*)carve((size_t)4096 * 4096 * sizeof(_Float16));
  _Float16* fc2p = (_Float16*)carve((size_t)SNN_LABELS * 4096 * sizeof(_Float16));

  // ---- zero state; prepack weights; convert input ----
  zero_f32<<<1, 128, 0, stream>>>((float*)zeroPix, 256);
  zero_f32<<<(int)((memTotal + 255) / 256), 256, 0, stream>>>(memBase, (int)memTotal);
  zero_f32<<<1, 64, 0, stream>>>(out, out_size);

  pack_conv0_w<<<(64 * 32 + 255) / 256, 256, 0, stream>>>(wconv[0], wpack[0]);
  for (int i = 1; i < 13; ++i) {
    int tot = cfg[i] * 9 * cfg[i - 1];
    pack_conv_w<<<(tot + 255) / 256, 256, 0, stream>>>(wconv[i], wpack[i],
                                                       cfg[i], cfg[i - 1]);
  }
  pack_fc0_w<<<(4096 * 2048 + 255) / 256, 256, 0, stream>>>(fc0w, fc0p);
  cvt_f32_f16<<<(4096 * 4096 + 255) / 256, 256, 0, stream>>>(fc1w, fc1p,
                                                             4096 * 4096);
  cvt_f32_f16<<<(SNN_LABELS * 4096 + 255) / 256, 256, 0, stream>>>(
      fc2w, fc2p, SNN_LABELS * 4096);
  {
    int tot = Bn * 1024 * 3;
    input_to_cl_f16<<<(tot + 255) / 256, 256, 0, stream>>>(x, Xf16, Bn);
  }

  // ---- T timesteps, sequential on `stream` ----
  _Float16* bufs[2] = {actA, actB};
  for (int t = 0; t < SNN_T; ++t) {
    conv0_lif_wmma<<<dim3(Bn * 1024 / 16, 1), 128, 0, stream>>>(
        wpack[0], Xf16, actA, memPtr[0], thr, leak, Bn);
    const _Float16* cur = actA;
    int tog = 1;
    int s   = 32;
    for (int i = 1; i < 13; ++i) {
      _Float16* outb = bufs[tog];
      const int NP   = Bn * s * s;
      const int cout = cfg[i], cin = cfg[i - 1];
      dim3 grid((NP + 31) / 32, cout / 64);
      switch (cin) {
        case 64:
          conv_lif_wmma<64><<<grid, 128, 0, stream>>>(wpack[i], cur, zeroPix,
              outb, memPtr[i], thr, leak, i, cout, s, Bn);
          break;
        case 128:
          conv_lif_wmma<128><<<grid, 128, 0, stream>>>(wpack[i], cur, zeroPix,
              outb, memPtr[i], thr, leak, i, cout, s, Bn);
          break;
        case 256:
          conv_lif_wmma<256><<<grid, 128, 0, stream>>>(wpack[i], cur, zeroPix,
              outb, memPtr[i], thr, leak, i, cout, s, Bn);
          break;
        default:
          conv_lif_wmma<512><<<grid, 128, 0, stream>>>(wpack[i], cur, zeroPix,
              outb, memPtr[i], thr, leak, i, cout, s, Bn);
          break;
      }
      cur = outb;
      tog ^= 1;
      if (i == 1 || i == 3 || i == 6 || i == 9) {
        _Float16* pb = bufs[tog];
        int tot = cfg[i] * Bn * (s / 2) * (s / 2);
        avgpool2_cl_f16<<<(tot + 255) / 256, 256, 0, stream>>>(cur, pb,
                                                               cfg[i], Bn, s);
        s >>= 1;
        cur = pb;
        tog ^= 1;
      }
    }
    // cur = conv12 output, channel-last [B][2][2][512] = [b][2048] contiguous
    fc_lif_wmma<<<dim3(1, 64), 128, 0, stream>>>(fc0p, cur, fcA, memPtr[13],
                                                 thr, leak, 13, 2048, 4096, Bn);
    fc_lif_wmma<<<dim3(1, 64), 128, 0, stream>>>(fc1p, fcA, fcB, memPtr[14],
                                                 thr, leak, 14, 4096, 4096, Bn);
    fc2_logits_wmma<<<1, 32, 0, stream>>>(fc2p, fcB, out, 4096, SNN_LABELS, Bn);
  }
}
